// SoftHd_74964359184574
// MI455X (gfx1250) — matrix operand order repaired and checked by reading, not compiled, not run on hardware
//
#include <hip/hip_runtime.h>

typedef __attribute__((ext_vector_type(2)))  float  v2f;
typedef __attribute__((ext_vector_type(8)))  float  v8f;
typedef __attribute__((ext_vector_type(4)))  __bf16 v4bf;
typedef __attribute__((ext_vector_type(8)))  __bf16 v8bf;
typedef __attribute__((ext_vector_type(16))) __bf16 v16bf;

#define B_ 32
#define N_ 1024
#define M_ 1024
#define D_ 256

#define BN 128
#define BM 128
#define BK 32
#define LDB (BK + 8) // LDS row stride in bf16 elems: 40*2 = 80B, 16B-aligned rows

// ---------------------------------------------------------------------------
// Kernel 1: per-row  sq = ||h||^2  and  min_init = |h . W + bias|
// One wave (32 lanes) per row of 256 floats; butterfly reduce in-wave.
// ---------------------------------------------------------------------------
__global__ __launch_bounds__(256) void softhd_prep(
    const float* __restrict__ h, const float* __restrict__ W,
    const float* __restrict__ bias, float* __restrict__ sq_out,
    float* __restrict__ min_out, int rows) {
  int gtid = blockIdx.x * blockDim.x + threadIdx.x;
  int row = gtid >> 5;
  int lane = gtid & 31;
  if (row >= rows) return;

  const float4* hp = (const float4*)(h + (size_t)row * D_);
  const float4* wp = (const float4*)W;
  float dot = 0.f, sq = 0.f;
#pragma unroll
  for (int it = 0; it < 2; ++it) {
    float4 x = hp[lane + it * 32];
    float4 w = wp[lane + it * 32];
    dot = fmaf(x.x, w.x, fmaf(x.y, w.y, fmaf(x.z, w.z, fmaf(x.w, w.w, dot))));
    sq  = fmaf(x.x, x.x, fmaf(x.y, x.y, fmaf(x.z, x.z, fmaf(x.w, x.w, sq))));
  }
#pragma unroll
  for (int off = 16; off; off >>= 1) {
    dot += __shfl_xor(dot, off, 32);
    sq  += __shfl_xor(sq, off, 32);
  }
  if (lane == 0) {
    sq_out[row]  = sq;
    min_out[row] = fabsf(dot + bias[0]);  // >= 0, so uint-bit atomicMin is valid
  }
}

// round-to-nearest-even f32 -> bf16 and back (native clang __bf16 conversions)
__device__ __forceinline__ __bf16 f2bf(float x) { return (__bf16)x; }
__device__ __forceinline__ float  bf2f(__bf16 b) { return (float)b; }

// WGP-scope prefetch (scope 0 = default): fetches into all cache levels on
// a miss.  __builtin_prefetch only produces SE/SYS scope, which skips the
// WGP-level prefetch entirely (ISA 10.5), so emit the instruction directly.
// Fire-and-forget: no VGPR return, no counter change.
__device__ __forceinline__ void prefetch_wgp(const void* p) {
  asm volatile("global_prefetch_b8 %0, off" ::"v"(p) : "memory");
}

// ---------------------------------------------------------------------------
// Kernel 2: batched cross-GEMM on the bf16 matrix pipe with split-precision
// (x = hi + lo; cross ~= hi*hi + hi*lo + lo*hi, f32 accumulate), fused
// distance epilogue + row/col min reduction via global atomic-min on bits.
// Grid: (M/128, N/128, B). Block: 256 threads = 8 waves (wave32).
// Wave grid: 4 row-waves x 2 col-waves; wave tile = 32n x 64m = 2x4 v8f accs.
// ---------------------------------------------------------------------------
__global__ __launch_bounds__(256) void softhd_main(
    const float* __restrict__ h1, const float* __restrict__ h2,
    const float* __restrict__ sq1, const float* __restrict__ sq2,
    unsigned* __restrict__ rowmin, unsigned* __restrict__ colmin) {
  __shared__ __align__(16) __bf16 Ahi[BN * LDB];
  __shared__ __align__(16) __bf16 Alo[BN * LDB];
  __shared__ __align__(16) __bf16 Bhi[BM * LDB];
  __shared__ __align__(16) __bf16 Blo[BM * LDB];
  __shared__ float S1[BN];
  __shared__ float S2[BM];

  const int b = blockIdx.z;
  const int n0 = blockIdx.y * BN;
  const int m0 = blockIdx.x * BM;
  const int tid = threadIdx.x;
  const int lane = tid & 31;
  const int wid = tid >> 5;
  const int wrow = (wid & 3) * 32;   // wave's n-offset inside tile
  const int wcol = (wid >> 2) * 64;  // wave's m-offset inside tile
  const int half = lane >> 4;        // 0: lanes 0-15, 1: lanes 16-31
  const int l15 = lane & 15;

  const float* A_g = h1 + ((size_t)b * N_ + n0) * D_;
  const float* B_g = h2 + ((size_t)b * M_ + m0) * D_;

  if (tid < BN) S1[tid] = sq1[b * N_ + n0 + tid];
  else          S2[tid - BN] = sq2[b * M_ + m0 + (tid - BN)];

  v8f acc[2][4] = {};

  const int r_ld = tid >> 3;          // 0..31
  const int c4 = (tid & 7) * 4;       // 0..28 step 4

  for (int k0 = 0; k0 < D_; k0 += BK) {
    // ---- Stage 128x32 f32 tiles, splitting each value into bf16 hi+lo ----
#pragma unroll
    for (int it = 0; it < 4; ++it) {
      const int row = r_ld + it * 32;
      float4 va = *(const float4*)(A_g + (size_t)row * D_ + k0 + c4);
      float4 vb = *(const float4*)(B_g + (size_t)row * D_ + k0 + c4);
      if (k0 + BK < D_) {  // warm WGP$ for the next K-chunk
        prefetch_wgp(A_g + (size_t)row * D_ + k0 + BK + c4);
        prefetch_wgp(B_g + (size_t)row * D_ + k0 + BK + c4);
      }
      v4bf ah, al, bh, bl;
      ah[0] = f2bf(va.x); al[0] = f2bf(va.x - bf2f(ah[0]));
      ah[1] = f2bf(va.y); al[1] = f2bf(va.y - bf2f(ah[1]));
      ah[2] = f2bf(va.z); al[2] = f2bf(va.z - bf2f(ah[2]));
      ah[3] = f2bf(va.w); al[3] = f2bf(va.w - bf2f(ah[3]));
      bh[0] = f2bf(vb.x); bl[0] = f2bf(vb.x - bf2f(bh[0]));
      bh[1] = f2bf(vb.y); bl[1] = f2bf(vb.y - bf2f(bh[1]));
      bh[2] = f2bf(vb.z); bl[2] = f2bf(vb.z - bf2f(bh[2]));
      bh[3] = f2bf(vb.w); bl[3] = f2bf(vb.w - bf2f(bh[3]));
      *(v4bf*)&Ahi[row * LDB + c4] = ah;
      *(v4bf*)&Alo[row * LDB + c4] = al;
      *(v4bf*)&Bhi[row * LDB + c4] = bh;
      *(v4bf*)&Blo[row * LDB + c4] = bl;
    }
    __syncthreads();

    // ---- Fragment loads (contiguous K order, same for A and B) ----
    // A 16x32: lane=row; 16 contiguous K starting at half*16.
    // B 32x16: lane=col; 16 contiguous K starting at half*16.
    v16bf fah[2], fal[2], fbh[4], fbl[4];
#pragma unroll
    for (int i = 0; i < 2; ++i) {
      const int row = wrow + 16 * i + l15;
      v8bf p0 = *(const v8bf*)&Ahi[row * LDB + half * 16];
      v8bf p1 = *(const v8bf*)&Ahi[row * LDB + half * 16 + 8];
      fah[i] = __builtin_shufflevector(p0, p1, 0, 1, 2, 3, 4, 5, 6, 7, 8, 9,
                                       10, 11, 12, 13, 14, 15);
      v8bf q0 = *(const v8bf*)&Alo[row * LDB + half * 16];
      v8bf q1 = *(const v8bf*)&Alo[row * LDB + half * 16 + 8];
      fal[i] = __builtin_shufflevector(q0, q1, 0, 1, 2, 3, 4, 5, 6, 7, 8, 9,
                                       10, 11, 12, 13, 14, 15);
    }
#pragma unroll
    for (int j = 0; j < 4; ++j) {
      const int col = wcol + 16 * j + l15;
      v8bf p0 = *(const v8bf*)&Bhi[col * LDB + half * 16];
      v8bf p1 = *(const v8bf*)&Bhi[col * LDB + half * 16 + 8];
      fbh[j] = __builtin_shufflevector(p0, p1, 0, 1, 2, 3, 4, 5, 6, 7, 8, 9,
                                       10, 11, 12, 13, 14, 15);
      v8bf q0 = *(const v8bf*)&Blo[col * LDB + half * 16];
      v8bf q1 = *(const v8bf*)&Blo[col * LDB + half * 16 + 8];
      fbl[j] = __builtin_shufflevector(q0, q1, 0, 1, 2, 3, 4, 5, 6, 7, 8, 9,
                                       10, 11, 12, 13, 14, 15);
    }

    // ---- 24 WMMAs: hi*hi + hi*lo + lo*hi (lo*lo ~ 2^-18 rel, dropped) ----
#pragma unroll
    for (int i = 0; i < 2; ++i) {
#pragma unroll
      for (int j = 0; j < 4; ++j) {
        acc[i][j] = __builtin_amdgcn_wmma_f32_16x16x32_bf16(
            false, fah[i], false, fbh[j], (short)0, acc[i][j], false, false);
        acc[i][j] = __builtin_amdgcn_wmma_f32_16x16x32_bf16(
            false, fah[i], false, fbl[j], (short)0, acc[i][j], false, false);
        acc[i][j] = __builtin_amdgcn_wmma_f32_16x16x32_bf16(
            false, fal[i], false, fbh[j], (short)0, acc[i][j], false, false);
      }
    }
    __syncthreads();
  }

  // Epilogue: dist = max(||x||^2 + ||y||^2 - 2*cross, 0) * 0.5 (in place).
  // C/D layout: VGPR v, lane l -> (n = v + 8*(l>>4), m = l&15) of 16x16 tile.
#pragma unroll
  for (int i = 0; i < 2; ++i) {
#pragma unroll
    for (int v = 0; v < 8; ++v) {
      const float s1 = S1[wrow + 16 * i + 8 * half + v];
#pragma unroll
      for (int j = 0; j < 4; ++j) {
        const float s2 = S2[wcol + 16 * j + l15];
        const float c = acc[i][j][v];
        acc[i][j][v] = fmaxf(s1 + s2 - 2.0f * c, 0.0f) * 0.5f;
      }
    }
  }

  // Row mins: min over this wave's 64 columns, then atomic-min to global.
#pragma unroll
  for (int i = 0; i < 2; ++i) {
#pragma unroll
    for (int v = 0; v < 8; ++v) {
      float r = fminf(fminf(acc[i][0][v], acc[i][1][v]),
                      fminf(acc[i][2][v], acc[i][3][v]));
      r = fminf(r, __shfl_xor(r, 1, 32));
      r = fminf(r, __shfl_xor(r, 2, 32));
      r = fminf(r, __shfl_xor(r, 4, 32));
      r = fminf(r, __shfl_xor(r, 8, 32));
      if (l15 == 0) {
        const int n = n0 + wrow + 16 * i + 8 * half + v;
        atomicMin(&rowmin[b * N_ + n], __float_as_uint(r));
      }
    }
  }

  // Col mins: min over this wave's 32 rows (8 VGPRs x 2 frags x 2 halves).
#pragma unroll
  for (int j = 0; j < 4; ++j) {
    float c = 3.402823466e+38f;
#pragma unroll
    for (int i = 0; i < 2; ++i)
#pragma unroll
      for (int v = 0; v < 8; ++v) c = fminf(c, acc[i][j][v]);
    c = fminf(c, __shfl_xor(c, 16, 32));
    if (half == 0) {
      const int m = m0 + wcol + 16 * j + l15;
      atomicMin(&colmin[b * M_ + m], __float_as_uint(c));
    }
  }
}

// ---------------------------------------------------------------------------
// Kernel 3: d[b] = mean(rowmin[b,:]) + mean(colmin[b,:])
// ---------------------------------------------------------------------------
__global__ __launch_bounds__(256) void softhd_final(
    const float* __restrict__ rowmin, const float* __restrict__ colmin,
    float* __restrict__ out) {
  const int b = blockIdx.x;
  const int tid = threadIdx.x;
  float acc = 0.f;
  for (int i = tid; i < N_; i += 256) acc += rowmin[b * N_ + i] * (1.0f / N_);
  for (int i = tid; i < M_; i += 256) acc += colmin[b * M_ + i] * (1.0f / M_);
#pragma unroll
  for (int off = 16; off; off >>= 1) acc += __shfl_xor(acc, off, 32);
  __shared__ float red[8];
  if ((tid & 31) == 0) red[tid >> 5] = acc;
  __syncthreads();
  if (tid < 8) {
    float v = red[tid];
    v += __shfl_xor(v, 4, 32);
    v += __shfl_xor(v, 2, 32);
    v += __shfl_xor(v, 1, 32);
    if (tid == 0) out[b] = v;
  }
}

// ---------------------------------------------------------------------------
extern "C" void kernel_launch(void* const* d_in, const int* in_sizes, int n_in,
                              void* d_out, int out_size, void* d_ws,
                              size_t ws_size, hipStream_t stream) {
  const float* h1 = (const float*)d_in[0];    // [B,N,D]
  const float* h2 = (const float*)d_in[1];    // [B,M,D]
  const float* W = (const float*)d_in[2];     // [D]
  const float* bias = (const float*)d_in[3];  // [1]
  float* out = (float*)d_out;                 // [B]

  float* rowmin = (float*)d_ws;          // B*N floats (init = d1, then min)
  float* colmin = rowmin + B_ * N_;      // B*M floats (init = d2, then min)
  float* sq1 = colmin + B_ * M_;         // B*N floats
  float* sq2 = sq1 + B_ * N_;            // B*M floats

  const int rows1 = B_ * N_;
  const int rows2 = B_ * M_;
  // 8 waves per 256-thread block, one row per wave.
  softhd_prep<<<rows1 / 8, 256, 0, stream>>>(h1, W, bias, sq1, rowmin, rows1);
  softhd_prep<<<rows2 / 8, 256, 0, stream>>>(h2, W, bias, sq2, colmin, rows2);

  dim3 grid(M_ / BM, N_ / BN, B_);
  softhd_main<<<grid, 256, 0, stream>>>(h1, h2, sq1, sq2, (unsigned*)rowmin,
                                        (unsigned*)colmin);

  softhd_final<<<B_, 256, 0, stream>>>(rowmin, colmin, out);
}
